// AdvancedLLM_35631048687821
// MI455X (gfx1250) — compile-verified, matrix operand order checked by reading, not run on hardware
//
#include <hip/hip_runtime.h>
#include <hip/hip_bf16.h>

#define D_MODEL 1024
#define D_FF    4096
#define SEQ     1024
#define NTOK    2048   // BATCH*SEQ
#define N_EXP   8
#define D_K     64

typedef __bf16 v16bf __attribute__((ext_vector_type(16)));
typedef float  v8f   __attribute__((ext_vector_type(8)));
typedef unsigned int u32x4 __attribute__((ext_vector_type(4)));
typedef int i32x4 __attribute__((ext_vector_type(4)));
typedef int i32x8 __attribute__((ext_vector_type(8)));

union FragU { v16bf v; uint4 q[2]; unsigned short s[16]; };

__device__ __forceinline__ unsigned short f2bf(float f) {
    union { float f; unsigned u; } a; a.f = f;
    unsigned r = a.u + 0x7FFFu + ((a.u >> 16) & 1u);
    return (unsigned short)(r >> 16);
}

// 16x32 bf16 fragment load from a row-major [16][stride] LDS tile (A layout;
// also B when the tile is stored N-major / B^T).  CDNA5 ISA 7.12.2 layout.
__device__ __forceinline__ v16bf load_frag(const unsigned short* base, int stride, int kofs) {
    int lane = threadIdx.x & 31;
    int row  = lane & 15;
    int kb   = (lane >> 4) << 3;
    const unsigned short* p = base + row * stride + kofs + kb;
    FragU u;
    u.q[0] = *(const uint4*)(p);
    u.q[1] = *(const uint4*)(p + 16);
    return u.v;
}

// ---------------------------------------------------------------------------
// Tensor Data Mover: DMA a 2D bf16 tile (tile_rows x 32 elems) from global to
// LDS, padding 4 dwords after every 16 dwords -> LDS row stride 40 ushorts
// (conflict-free for the fragment loader).  D# layout per CDNA5 ISA ch.8.
// amdgpu-toolchain 6-arg builtin form.
// ---------------------------------------------------------------------------
__device__ __forceinline__ void tdm_load_tile(unsigned lds_addr, const unsigned short* gptr,
                                              int tile_rows, long row_stride_elems)
{
    unsigned long long ga = (unsigned long long)(size_t)gptr;
    u32x4 g0;
    g0[0] = 1u;                                              // count=1, user mode
    g0[1] = lds_addr;                                        // lds_addr[31:0]
    g0[2] = (unsigned)(ga & 0xFFFFFFFFu);                    // global_addr[31:0]
    g0[3] = (unsigned)((ga >> 32) & 0x01FFFFFFu) | (2u << 30); // addr[56:32] | type=2
    i32x8 g1;
    g1[0] = (1 << 16) | (1 << 20) | (3 << 22) | (3 << 25);   // data_size=2B, pad_en, int=16dw, amt=4dw
    g1[1] = 0;                                               // tensor_dim0[15:0]<<16 (=0, dim0 big below)
    g1[2] = 0x4000;                                          // tensor_dim0 = 1<<30 ; dim1 low16=0
    g1[3] = 0x4000 | (32 << 16);                             // tensor_dim1 = 1<<30 ; tile_dim0=32
    g1[4] = tile_rows & 0xFFFF;                              // tile_dim1 ; tile_dim2=0
    g1[5] = (int)(row_stride_elems & 0xFFFFFFFF);            // tensor_dim0_stride[31:0]
    g1[6] = (int)((row_stride_elems >> 32) & 0xFFFF);        // stride[47:32] ; dim1_stride=0
    g1[7] = 0;
    i32x4 gz4 = {0, 0, 0, 0};
    i32x8 gz8 = {0, 0, 0, 0, 0, 0, 0, 0};
    __builtin_amdgcn_tensor_load_to_lds(g0, g1, gz4, gz4, gz8, 0);
}

// ---------------------------------------------------------------------------
// Unified bf16 WMMA GEMM, TDM-staged, double buffered.
// C[M,N] = A[M,K](bf16) * BT[N,K](bf16)^T  (+bias)(+resid), fp32 out.
// Per-z batched BT/bias; per-z A/C row ranges via offsets[z]..offsets[z+1].
// ---------------------------------------------------------------------------
__global__ __launch_bounds__(256) void gemm_tdm(
    const unsigned short* __restrict__ A, const unsigned short* __restrict__ BT,
    float* __restrict__ C,
    int M, int N, int K, int lda, int ldbt, int ldc,
    long bzstride, const float* __restrict__ bias, long biaszstride,
    const float* __restrict__ resid, const int* __restrict__ offsets)
{
    __shared__ __align__(16) unsigned short As[2][128 * 40];
    __shared__ __align__(16) unsigned short Bs[2][128 * 40];

    int z    = blockIdx.z;
    int ofs  = offsets ? offsets[z] : 0;
    int Meff = offsets ? (offsets[z + 1] - ofs) : M;
    int mbase = blockIdx.x * 128;
    if (mbase >= Meff) return;
    int nbase = blockIdx.y * 128;
    const unsigned short* Az  = A + (long)(ofs + mbase) * lda;
    const unsigned short* BTz = BT + (long)z * bzstride + (long)nbase * ldbt;

    int tid = threadIdx.x;
    int w   = tid >> 5;
    int wm  = (w & 1) * 64;
    int wn  = (w >> 1) * 32;

    v8f acc[4][2];
#pragma unroll
    for (int i = 0; i < 4; ++i)
#pragma unroll
        for (int j = 0; j < 2; ++j) acc[i][j] = (v8f){};

    int nk = K / 32;
    if (w == 0) {
        tdm_load_tile((unsigned)(size_t)&As[0][0], Az, 128, lda);
        tdm_load_tile((unsigned)(size_t)&Bs[0][0], BTz, 128, ldbt);
        __builtin_amdgcn_s_wait_tensorcnt(0);
    }
    __syncthreads();

    for (int it = 0; it < nk; ++it) {
        int cur = it & 1;
        if (w == 0 && it + 1 < nk) {
            int k0 = (it + 1) * 32;
            tdm_load_tile((unsigned)(size_t)&As[cur ^ 1][0], Az + k0, 128, lda);
            tdm_load_tile((unsigned)(size_t)&Bs[cur ^ 1][0], BTz + k0, 128, ldbt);
        }
#pragma unroll
        for (int i = 0; i < 4; ++i) {
            v16bf a = load_frag(&As[cur][0] + (wm + i * 16) * 40, 40, 0);
#pragma unroll
            for (int j = 0; j < 2; ++j) {
                v16bf b = load_frag(&Bs[cur][0] + (wn + j * 16) * 40, 40, 0);
                acc[i][j] = __builtin_amdgcn_wmma_f32_16x16x32_bf16(
                    false, a, false, b, (short)0, acc[i][j], false, false);
            }
        }
        __syncthreads();                       // all waves done reading buf[cur]
        if (w == 0 && it + 1 < nk)
            __builtin_amdgcn_s_wait_tensorcnt(0);
        __syncthreads();                       // buf[cur^1] published
    }

    int lane = tid & 31;
    int col0 = nbase + wn + (lane & 15);
#pragma unroll
    for (int i = 0; i < 4; ++i)
#pragma unroll
        for (int j = 0; j < 2; ++j)
#pragma unroll
            for (int r = 0; r < 8; ++r) {
                int lr   = wm + i * 16 + ((lane < 16) ? r : r + 8);
                int grow = mbase + lr;
                if (grow < Meff) {
                    int col = col0 + j * 16;
                    float v = acc[i][j][r];
                    if (bias) v += bias[(long)z * biaszstride + col];
                    long cidx = (long)(ofs + grow) * ldc + col;
                    if (resid) v += resid[cidx];
                    C[cidx] = v;
                }
            }
}

// ---------------------------------------------------------------------------
// Fused MoE up-projection: h = (x*W2+b2) * silu(x*W1+b1), bf16 out.
// A rows gathered by token list (VALU copy); W1T/W2T tiles staged via TDM.
// BM=128, BN=64. grid = (Mtiles, DFF/64, 8 experts)
// ---------------------------------------------------------------------------
__global__ __launch_bounds__(256) void moe_h12(
    const unsigned short* __restrict__ X, const unsigned short* __restrict__ W1T,
    const unsigned short* __restrict__ W2T,
    const float* __restrict__ b1, const float* __restrict__ b2,
    unsigned short* __restrict__ H,
    const int* __restrict__ tokens, const int* __restrict__ offsets)
{
    __shared__ __align__(16) unsigned short As[128 * 40];
    __shared__ __align__(16) unsigned short B1s[64 * 40];
    __shared__ __align__(16) unsigned short B2s[64 * 40];

    int z    = blockIdx.z;
    int ofs  = offsets[z];
    int Meff = offsets[z + 1] - ofs;
    int mbase = blockIdx.x * 128;
    if (mbase >= Meff) return;
    int nbase = blockIdx.y * 64;
    const unsigned short* B1z = W1T + (long)z * D_MODEL * D_FF + (long)nbase * D_MODEL;
    const unsigned short* B2z = W2T + (long)z * D_MODEL * D_FF + (long)nbase * D_MODEL;

    int tid = threadIdx.x;
    int w   = tid >> 5;
    int wm  = (w & 1) * 64;
    int wn  = (w >> 1) * 16;

    v8f a1[4], a2[4];
#pragma unroll
    for (int i = 0; i < 4; ++i) { a1[i] = (v8f){}; a2[i] = (v8f){}; }

    for (int k0 = 0; k0 < D_MODEL; k0 += 32) {
        if (w == 0) {
            tdm_load_tile((unsigned)(size_t)&B1s[0], B1z + k0, 64, D_MODEL);
            tdm_load_tile((unsigned)(size_t)&B2s[0], B2z + k0, 64, D_MODEL);
        }
        // gather A rows (bf16 copy, 8-ushort chunks)
#pragma unroll
        for (int i = 0; i < 2; ++i) {
            int id = tid + i * 256;
            int r  = id >> 2;
            int c8 = (id & 3) * 8;
            uint4 v = {};
            int gr = mbase + r;
            if (gr < Meff) {
                int ar = tokens[ofs + gr];
                v = *(const uint4*)(X + (long)ar * D_MODEL + k0 + c8);
            }
            *(uint4*)(As + r * 40 + c8) = v;
        }
        if (w == 0) __builtin_amdgcn_s_wait_tensorcnt(0);
        __syncthreads();
        v16bf bf1 = load_frag(B1s + wn * 40, 40, 0);
        v16bf bf2 = load_frag(B2s + wn * 40, 40, 0);
#pragma unroll
        for (int i = 0; i < 4; ++i) {
            v16bf a = load_frag(As + (wm + i * 16) * 40, 40, 0);
            a1[i] = __builtin_amdgcn_wmma_f32_16x16x32_bf16(false, a, false, bf1, (short)0, a1[i], false, false);
            a2[i] = __builtin_amdgcn_wmma_f32_16x16x32_bf16(false, a, false, bf2, (short)0, a2[i], false, false);
        }
        __syncthreads();
    }

    int lane = tid & 31;
    int col  = nbase + wn + (lane & 15);
    float bb1 = b1[(long)z * D_FF + col];
    float bb2 = b2[(long)z * D_FF + col];
#pragma unroll
    for (int i = 0; i < 4; ++i)
#pragma unroll
        for (int r = 0; r < 8; ++r) {
            int lr   = wm + i * 16 + ((lane < 16) ? r : r + 8);
            int grow = mbase + lr;
            if (grow < Meff) {
                float v1 = a1[i][r] + bb1;
                float v2 = a2[i][r] + bb2;
                float hval = v2 * (v1 / (1.0f + __expf(-v1)));
                H[(long)(ofs + grow) * D_FF + col] = f2bf(hval);
            }
        }
}

// ---------------------------------------------------------------------------
// Flash-style causal GQA attention (fp32 in, bf16 out). grid=(S/64, B*H), 128 thr.
// ---------------------------------------------------------------------------
__global__ __launch_bounds__(128) void flash_attn(
    const float* __restrict__ Q, const float* __restrict__ Kb,
    const float* __restrict__ Vb, unsigned short* __restrict__ O)
{
    __shared__ unsigned short Qs[64 * 72];
    __shared__ unsigned short Ks[32 * 72];
    __shared__ unsigned short Vts[64 * 40];
    __shared__ unsigned short Ps[4][16 * 40];

    int bh = blockIdx.y;
    int b  = bh >> 4;
    int h  = bh & 15;
    int g  = h >> 2;
    int qbase = blockIdx.x * 64;
    int tid = threadIdx.x, w = tid >> 5, lane = tid & 31;

#pragma unroll
    for (int i = 0; i < 8; ++i) {
        int id = tid + i * 128;
        int r  = id >> 4;
        int c4 = (id & 15) * 4;
        float4 v = *(const float4*)(Q + (long)(b * SEQ + qbase + r) * D_MODEL + h * D_K + c4);
        unsigned short* d = Qs + r * 72 + c4;
        d[0] = f2bf(v.x); d[1] = f2bf(v.y); d[2] = f2bf(v.z); d[3] = f2bf(v.w);
    }

    v8f o[4];
#pragma unroll
    for (int i = 0; i < 4; ++i) o[i] = (v8f){};
    float mrow[8], lsum[8];
#pragma unroll
    for (int r = 0; r < 8; ++r) { mrow[r] = -3.0e38f; lsum[r] = 0.0f; }

    int nkt = blockIdx.x * 2 + 2;
    for (int kt = 0; kt < nkt; ++kt) {
        int tbase = kt * 32;
        __syncthreads();
#pragma unroll
        for (int i = 0; i < 4; ++i) {
            int id = tid + i * 128;
            int tr = id >> 4;
            int c4 = (id & 15) * 4;
            long src = (long)(b * SEQ + tbase + tr) * 256 + g * D_K + c4;
            float4 kv = *(const float4*)(Kb + src);
            unsigned short* d = Ks + tr * 72 + c4;
            d[0] = f2bf(kv.x); d[1] = f2bf(kv.y); d[2] = f2bf(kv.z); d[3] = f2bf(kv.w);
            float4 vv = *(const float4*)(Vb + src);
            Vts[(c4 + 0) * 40 + tr] = f2bf(vv.x);
            Vts[(c4 + 1) * 40 + tr] = f2bf(vv.y);
            Vts[(c4 + 2) * 40 + tr] = f2bf(vv.z);
            Vts[(c4 + 3) * 40 + tr] = f2bf(vv.w);
        }
        __syncthreads();

        v8f s0 = (v8f){}, s1 = (v8f){};
#pragma unroll
        for (int ks = 0; ks < 64; ks += 32) {
            v16bf a   = load_frag(Qs + (w * 16) * 72, 72, ks);
            v16bf b0  = load_frag(Ks, 72, ks);
            v16bf b1f = load_frag(Ks + 16 * 72, 72, ks);
            s0 = __builtin_amdgcn_wmma_f32_16x16x32_bf16(false, a, false, b0, (short)0, s0, false, false);
            s1 = __builtin_amdgcn_wmma_f32_16x16x32_bf16(false, a, false, b1f, (short)0, s1, false, false);
        }

        int colg = tbase + (lane & 15);
#pragma unroll
        for (int r = 0; r < 8; ++r) {
            int rowg = qbase + w * 16 + ((lane < 16) ? r : r + 8);
            float v0 = s0[r] * 0.125f;
            float v1 = s1[r] * 0.125f;
            if (colg > rowg) v0 = -3.0e38f;
            if (colg + 16 > rowg) v1 = -3.0e38f;
            float mx = fmaxf(v0, v1);
            mx = fmaxf(mx, __shfl_xor(mx, 1, 16));
            mx = fmaxf(mx, __shfl_xor(mx, 2, 16));
            mx = fmaxf(mx, __shfl_xor(mx, 4, 16));
            mx = fmaxf(mx, __shfl_xor(mx, 8, 16));
            float mnew  = fmaxf(mrow[r], mx);
            float alpha = __expf(mrow[r] - mnew);
            float p0 = __expf(v0 - mnew);
            float p1 = __expf(v1 - mnew);
            float ps = p0 + p1;
            ps += __shfl_xor(ps, 1, 16);
            ps += __shfl_xor(ps, 2, 16);
            ps += __shfl_xor(ps, 4, 16);
            ps += __shfl_xor(ps, 8, 16);
            lsum[r] = lsum[r] * alpha + ps;
            mrow[r] = mnew;
            o[0][r] *= alpha; o[1][r] *= alpha; o[2][r] *= alpha; o[3][r] *= alpha;
            int prow = (lane < 16) ? r : r + 8;
            Ps[w][prow * 40 + (lane & 15)]      = f2bf(p0);
            Ps[w][prow * 40 + 16 + (lane & 15)] = f2bf(p1);
        }
        v16bf pa = load_frag(Ps[w], 40, 0);
#pragma unroll
        for (int n4 = 0; n4 < 4; ++n4) {
            v16bf vbf = load_frag(Vts + n4 * 16 * 40, 40, 0);
            o[n4] = __builtin_amdgcn_wmma_f32_16x16x32_bf16(false, pa, false, vbf, (short)0, o[n4], false, false);
        }
    }

#pragma unroll
    for (int n4 = 0; n4 < 4; ++n4)
#pragma unroll
        for (int r = 0; r < 8; ++r) {
            int lr  = w * 16 + ((lane < 16) ? r : r + 8);
            float val = o[n4][r] / lsum[r];
            O[(long)(b * SEQ + qbase + lr) * D_MODEL + h * D_K + n4 * 16 + (lane & 15)] = f2bf(val);
        }
}

// ---------------------------------------------------------------------------
// Transpose-convert: src fp32 [K][N] -> dst bf16 [N][K]   (per-z batched)
// ---------------------------------------------------------------------------
__global__ __launch_bounds__(256) void transpose_bf16(
    const float* __restrict__ src, unsigned short* __restrict__ dst, int K, int N)
{
    __shared__ float t[32][33];
    long zofs = (long)blockIdx.z * K * N;
    int k0 = blockIdx.x * 32, n0 = blockIdx.y * 32;
    int tid = threadIdx.x;
    int r  = tid >> 3;
    int c4 = (tid & 7) * 4;
    float4 v = *(const float4*)(src + zofs + (long)(k0 + r) * N + n0 + c4);
    t[r][c4 + 0] = v.x; t[r][c4 + 1] = v.y; t[r][c4 + 2] = v.z; t[r][c4 + 3] = v.w;
    __syncthreads();
    unsigned short* d = dst + zofs + (long)(n0 + r) * K + k0 + c4;
#pragma unroll
    for (int j = 0; j < 4; ++j) d[j] = f2bf(t[c4 + j][r]);
}

// ---------------------------------------------------------------------------
// RMSNorm kernels
// ---------------------------------------------------------------------------
__global__ __launch_bounds__(256) void rmsnorm_rope(
    const float* __restrict__ x, const float* __restrict__ w, unsigned short* __restrict__ out)
{
    __shared__ float red[256];
    int t = blockIdx.x, s = t & (SEQ - 1), tid = threadIdx.x;
    const float* xp = x + (long)t * D_MODEL;
    float ss = 0.0f;
    for (int i = tid; i < D_MODEL; i += 256) { float v = xp[i]; ss += v * v; }
    red[tid] = ss; __syncthreads();
    for (int st = 128; st > 0; st >>= 1) { if (tid < st) red[tid] += red[tid + st]; __syncthreads(); }
    float scale = rsqrtf(red[0] * (1.0f / D_MODEL) + 1e-6f);
    unsigned short* op = out + (long)t * D_MODEL;
    for (int j = tid; j < D_MODEL / 2; j += 256) {
        float xe = xp[2 * j] * scale * w[2 * j];
        float xo = xp[2 * j + 1] * scale * w[2 * j + 1];
        float theta = __expf(-(float)j * (9.210340371976184f / (D_MODEL / 2)));
        float sn, cs;
        __sincosf((float)s * theta, &sn, &cs);
        op[j] = f2bf(xe * cs - xo * sn);
        op[D_MODEL / 2 + j] = f2bf(xe * sn + xo * cs);
    }
}

__global__ __launch_bounds__(256) void rmsnorm_plain(
    const float* __restrict__ x, const float* __restrict__ w,
    float* __restrict__ out, unsigned short* __restrict__ outb)
{
    __shared__ float red[256];
    int t = blockIdx.x, tid = threadIdx.x;
    const float* xp = x + (long)t * D_MODEL;
    float ss = 0.0f;
    for (int i = tid; i < D_MODEL; i += 256) { float v = xp[i]; ss += v * v; }
    red[tid] = ss; __syncthreads();
    for (int st = 128; st > 0; st >>= 1) { if (tid < st) red[tid] += red[tid + st]; __syncthreads(); }
    float scale = rsqrtf(red[0] * (1.0f / D_MODEL) + 1e-6f);
    for (int i = tid; i < D_MODEL; i += 256) {
        float v = xp[i] * scale * w[i];
        out[(long)t * D_MODEL + i] = v;
        outb[(long)t * D_MODEL + i] = f2bf(v);
    }
}

// ---------------------------------------------------------------------------
// Router top-2 + softmax weights
// ---------------------------------------------------------------------------
__global__ __launch_bounds__(256) void router_topk(
    const float* __restrict__ xh, const float* __restrict__ rw, const float* __restrict__ rb,
    float* __restrict__ topw, int* __restrict__ topi)
{
    __shared__ float red[N_EXP * 256];
    int t = blockIdx.x, tid = threadIdx.x;
    float acc[N_EXP];
#pragma unroll
    for (int e = 0; e < N_EXP; ++e) acc[e] = 0.0f;
    for (int i = tid; i < D_MODEL; i += 256) {
        float xv = xh[(long)t * D_MODEL + i];
        const float* r = rw + (long)i * N_EXP;
#pragma unroll
        for (int e = 0; e < N_EXP; ++e) acc[e] += xv * r[e];
    }
#pragma unroll
    for (int e = 0; e < N_EXP; ++e) red[e * 256 + tid] = acc[e];
    __syncthreads();
    for (int st = 128; st > 0; st >>= 1) {
        if (tid < st)
#pragma unroll
            for (int e = 0; e < N_EXP; ++e) red[e * 256 + tid] += red[e * 256 + tid + st];
        __syncthreads();
    }
    if (tid == 0) {
        float lg[N_EXP];
#pragma unroll
        for (int e = 0; e < N_EXP; ++e) lg[e] = red[e * 256] + rb[e];
        int i0 = 0;
        for (int e = 1; e < N_EXP; ++e) if (lg[e] > lg[i0]) i0 = e;
        int i1 = (i0 == 0) ? 1 : 0;
        for (int e = 0; e < N_EXP; ++e) if (e != i0 && lg[e] > lg[i1]) i1 = e;
        float ex = __expf(lg[i1] - lg[i0]);
        topi[t * 2] = i0; topi[t * 2 + 1] = i1;
        topw[t * 2] = 1.0f / (1.0f + ex);
        topw[t * 2 + 1] = ex / (1.0f + ex);
    }
}

// Deterministic per-expert compaction (no atomics)
__global__ void moe_compact(const int* __restrict__ topi, int* __restrict__ tokens,
                            int* __restrict__ pairslot, int* __restrict__ offsets)
{
    __shared__ int cnt[N_EXP];
    __shared__ int ofs_s[N_EXP + 1];
    int tid = threadIdx.x;
    if (tid < N_EXP) {
        int c = 0;
        for (int t = 0; t < NTOK; ++t) {
            if (topi[2 * t] == tid) c++;
            if (topi[2 * t + 1] == tid) c++;
        }
        cnt[tid] = c;
    }
    __syncthreads();
    if (tid == 0) {
        int s = 0;
        for (int e = 0; e < N_EXP; ++e) { ofs_s[e] = s; s += cnt[e]; }
        ofs_s[N_EXP] = s;
        for (int e = 0; e <= N_EXP; ++e) offsets[e] = ofs_s[e];
    }
    __syncthreads();
    if (tid < N_EXP) {
        int j = ofs_s[tid];
        for (int t = 0; t < NTOK; ++t)
            for (int k = 0; k < 2; ++k)
                if (topi[2 * t + k] == tid) { tokens[j] = t; pairslot[2 * t + k] = j; ++j; }
    }
}

__global__ __launch_bounds__(256) void moe_combine(
    const float* __restrict__ eo, const int* __restrict__ pairslot,
    const float* __restrict__ topw, float* __restrict__ out)
{
    int t = blockIdx.x, tid = threadIdx.x;
    int s0 = pairslot[2 * t], s1 = pairslot[2 * t + 1];
    float w0 = topw[2 * t], w1 = topw[2 * t + 1];
    for (int d = tid; d < D_MODEL; d += 256)
        out[(long)t * D_MODEL + d] += w0 * eo[(long)s0 * D_MODEL + d] + w1 * eo[(long)s1 * D_MODEL + d];
}

// ---------------------------------------------------------------------------
extern "C" void kernel_launch(void* const* d_in, const int* in_sizes, int n_in,
                              void* d_out, int out_size, void* d_ws, size_t ws_size,
                              hipStream_t stream) {
    const float* x       = (const float*)d_in[0];
    const float* norm1_w = (const float*)d_in[1];
    const float* Wq      = (const float*)d_in[2];
    const float* Wk      = (const float*)d_in[3];
    const float* Wv      = (const float*)d_in[4];
    const float* Wo      = (const float*)d_in[5];
    const float* norm2_w = (const float*)d_in[6];
    const float* rw      = (const float*)d_in[7];
    const float* rb      = (const float*)d_in[8];
    const float* W1      = (const float*)d_in[9];
    const float* b1      = (const float*)d_in[10];
    const float* W2      = (const float*)d_in[11];
    const float* b2      = (const float*)d_in[12];
    const float* W3      = (const float*)d_in[13];
    const float* b3      = (const float*)d_in[14];
    float* out = (float*)d_out;

    char* ws = (char*)d_ws;
    unsigned short* xr   = (unsigned short*)ws; ws += (size_t)NTOK * D_MODEL * 2;
    unsigned short* WqT  = (unsigned short*)ws; ws += (size_t)D_MODEL * D_MODEL * 2;
    unsigned short* WkT  = (unsigned short*)ws; ws += (size_t)256 * D_MODEL * 2;
    unsigned short* WvT  = (unsigned short*)ws; ws += (size_t)256 * D_MODEL * 2;
    unsigned short* WoT  = (unsigned short*)ws; ws += (size_t)D_MODEL * D_MODEL * 2;
    float* Qb   = (float*)ws; ws += (size_t)NTOK * D_MODEL * 4;
    float* Kbuf = (float*)ws; ws += (size_t)NTOK * 256 * 4;
    float* Vbuf = (float*)ws; ws += (size_t)NTOK * 256 * 4;
    unsigned short* attnb = (unsigned short*)ws; ws += (size_t)NTOK * D_MODEL * 2;
    float* xh   = (float*)ws; ws += (size_t)NTOK * D_MODEL * 4;
    unsigned short* xhb  = (unsigned short*)ws; ws += (size_t)NTOK * D_MODEL * 2;
    unsigned short* W1T  = (unsigned short*)ws; ws += (size_t)N_EXP * D_MODEL * D_FF * 2;
    unsigned short* W2T  = (unsigned short*)ws; ws += (size_t)N_EXP * D_MODEL * D_FF * 2;
    unsigned short* W3T  = (unsigned short*)ws; ws += (size_t)N_EXP * D_MODEL * D_FF * 2;
    unsigned short* hbuf = (unsigned short*)ws; ws += (size_t)(2 * NTOK + 128) * D_FF * 2;
    float* eo   = (float*)ws; ws += (size_t)2 * NTOK * D_MODEL * 4;
    float* topw = (float*)ws; ws += (size_t)NTOK * 2 * 4;
    int* topi     = (int*)ws; ws += (size_t)NTOK * 2 * 4;
    int* pairslot = (int*)ws; ws += (size_t)NTOK * 2 * 4;
    int* tokens   = (int*)ws; ws += (size_t)2 * NTOK * 4;
    int* offsets  = (int*)ws; ws += 16 * 4;

    // weight transpose+bf16 conversion (once per call)
    transpose_bf16<<<dim3(32, 32, 1), 256, 0, stream>>>(Wq, WqT, D_MODEL, D_MODEL);
    transpose_bf16<<<dim3(32, 8, 1), 256, 0, stream>>>(Wk, WkT, D_MODEL, 256);
    transpose_bf16<<<dim3(32, 8, 1), 256, 0, stream>>>(Wv, WvT, D_MODEL, 256);
    transpose_bf16<<<dim3(32, 32, 1), 256, 0, stream>>>(Wo, WoT, D_MODEL, D_MODEL);
    transpose_bf16<<<dim3(32, 128, N_EXP), 256, 0, stream>>>(W1, W1T, D_MODEL, D_FF);
    transpose_bf16<<<dim3(32, 128, N_EXP), 256, 0, stream>>>(W2, W2T, D_MODEL, D_FF);
    transpose_bf16<<<dim3(128, 32, N_EXP), 256, 0, stream>>>(W3, W3T, D_FF, D_MODEL);

    // 1. xr = rope(rmsnorm(x))  (bf16)
    rmsnorm_rope<<<NTOK, 256, 0, stream>>>(x, norm1_w, xr);
    // 2-4. Q/K/V projections (TDM-staged WMMA)
    gemm_tdm<<<dim3(16, 8, 1), 256, 0, stream>>>(xr, WqT, Qb, NTOK, D_MODEL, D_MODEL,
        D_MODEL, D_MODEL, D_MODEL, 0, nullptr, 0, nullptr, nullptr);
    gemm_tdm<<<dim3(16, 2, 1), 256, 0, stream>>>(xr, WkT, Kbuf, NTOK, 256, D_MODEL,
        D_MODEL, D_MODEL, 256, 0, nullptr, 0, nullptr, nullptr);
    gemm_tdm<<<dim3(16, 2, 1), 256, 0, stream>>>(xr, WvT, Vbuf, NTOK, 256, D_MODEL,
        D_MODEL, D_MODEL, 256, 0, nullptr, 0, nullptr, nullptr);
    // 5. causal flash attention (WMMA QK^T / PV)
    flash_attn<<<dim3(16, 32, 1), 128, 0, stream>>>(Qb, Kbuf, Vbuf, attnb);
    // 6. out = x + attn @ Wo
    gemm_tdm<<<dim3(16, 8, 1), 256, 0, stream>>>(attnb, WoT, out, NTOK, D_MODEL, D_MODEL,
        D_MODEL, D_MODEL, D_MODEL, 0, nullptr, 0, x, nullptr);
    // 7. xh = rmsnorm(out)  (fp32 + bf16 shadow)
    rmsnorm_plain<<<NTOK, 256, 0, stream>>>(out, norm2_w, xh, xhb);
    // 8. router top-2
    router_topk<<<NTOK, 256, 0, stream>>>(xh, rw, rb, topw, topi);
    // 9. deterministic compaction
    moe_compact<<<1, 32, 0, stream>>>(topi, tokens, pairslot, offsets);
    // 10. fused expert up-projection + silu gate (gathered A, TDM B)
    moe_h12<<<dim3(16, D_FF / 64, N_EXP), 256, 0, stream>>>(xhb, W1T, W2T, b1, b2, hbuf, tokens, offsets);
    // 11. expert down-projection (+b3), per-expert row ranges
    gemm_tdm<<<dim3(16, 8, N_EXP), 256, 0, stream>>>(hbuf, W3T, eo, NTOK, D_MODEL, D_FF,
        D_FF, D_FF, D_MODEL, (long)D_FF * D_MODEL, b3, D_MODEL, nullptr, offsets);
    // 12. weighted combine into residual
    moe_combine<<<NTOK, 256, 0, stream>>>(eo, pairslot, topw, out);
}
